// AttenDecoder_38620345926279
// MI455X (gfx1250) — compile-verified, matrix operand order checked by reading, not run on hardware
//
#include <hip/hip_runtime.h>
#include <hip/hip_bf16.h>
#include <cstdint>

// Problem constants (match reference)
#define B_   64
#define S_   128
#define T_   32
#define H_   512
#define E_   512
#define VOC_ 32000
#define A_   1024   // 2H

typedef __attribute__((ext_vector_type(16))) __bf16 v16bf;
typedef __attribute__((ext_vector_type(8)))  __bf16 v8bf;
typedef __attribute__((ext_vector_type(8)))  float  v8f;

// ---------------- helpers ----------------
__device__ __forceinline__ uint16_t f32_to_bf16(float f) {
    union { float f; uint32_t u; } v; v.f = f;
    uint32_t u = v.u;
    u += 0x7FFFu + ((u >> 16) & 1u);   // round-to-nearest-even
    return (uint16_t)(u >> 16);
}

__device__ __forceinline__ float sigm(float x) { return 1.0f / (1.0f + __expf(-x)); }

// Load 16-bit A/B WMMA fragment from row-major [rows, ld] (ld = K): the
// CDNA5 16-bit 16x32 layout: lanes 0-15 hold K 0..7,16..23; lanes 16-31 hold
// K 8..15,24..31 for row = lane%16.  Two aligned 16B loads per lane.
__device__ __forceinline__ v16bf load_frag_nk(const uint16_t* base, int row, int ld,
                                              int kb, int lane) {
    int off = (lane < 16) ? 0 : 8;
    const __bf16* p = reinterpret_cast<const __bf16*>(base) + (size_t)row * ld + kb + off;
    v8bf lo = *reinterpret_cast<const v8bf*>(p);
    v8bf hi = *reinterpret_cast<const v8bf*>(p + 16);
    v16bf r;
#pragma unroll
    for (int i = 0; i < 8; ++i) { r[i] = lo[i]; r[8 + i] = hi[i]; }
    return r;
}

// Same fragment but source is row-major [K, N] (ld = N): gather loads.
__device__ __forceinline__ v16bf load_frag_kn(const uint16_t* base, int n, int ld,
                                              int kb, int lane) {
    int off = (lane < 16) ? 0 : 8;
    const __bf16* p = reinterpret_cast<const __bf16*>(base);
    v16bf r;
#pragma unroll
    for (int e = 0; e < 16; ++e) {
        int k = kb + off + ((e < 8) ? e : (8 + e));
        r[e] = p[(size_t)k * ld + n];
    }
    return r;
}

// ---------------- WMMA GEMM ----------------
// C[M,N] (fp32, row-major, ldc) = A[M,K] (bf16 row-major) x W^T  (+bias)
// BKN=false: W stored row-major [N,K]  (torch Linear weight)
// BKN=true : B stored row-major [K,N]
// Block = 128 threads = 4 waves.  Each wave computes a 64 x (16*NT) output
// tile: 4 M-tiles (m strip blockIdx.y*64) x NT N-tiles, 4*NT accumulators.
// Per K-step: NT B-frags + 4 A-frags -> 4*NT WMMAs (NT=4: 16 WMMA / 8 loads).
template <bool BKN, int NT>
__global__ void gemm_wmma_bf16(const uint16_t* __restrict__ A, long long aBatch, int lda,
                               const uint16_t* __restrict__ Bm, long long bBatch, int ldb,
                               const float* __restrict__ bias,
                               float* __restrict__ C, long long cBatch, int ldc,
                               int N, int K) {
    const int lane = threadIdx.x & 31;
    const int wave = threadIdx.x >> 5;
    const int n0 = (blockIdx.x * 4 + wave) * (16 * NT);
    if (n0 >= N) return;
    const int m0 = blockIdx.y * 64;
    const uint16_t* Ab = A + (size_t)blockIdx.z * aBatch;
    const uint16_t* Bb = Bm + (size_t)blockIdx.z * bBatch;
    float* Cb = C + (size_t)blockIdx.z * cBatch;

    const int rA = m0 + (lane & 15);          // A row for this lane (+ 16*i)
    const int cn = n0 + (lane & 15);          // first output column (+ 16*j)

    v8f acc[4][NT];
#pragma unroll
    for (int i = 0; i < 4; ++i)
#pragma unroll
        for (int j = 0; j < NT; ++j) acc[i][j] = v8f{};

    for (int kb = 0; kb < K; kb += 32) {
        v16bf bf[NT];
#pragma unroll
        for (int j = 0; j < NT; ++j)
            bf[j] = BKN ? load_frag_kn(Bb, cn + 16 * j, ldb, kb, lane)
                        : load_frag_nk(Bb, cn + 16 * j, ldb, kb, lane);
        v16bf af[4];
#pragma unroll
        for (int i = 0; i < 4; ++i)
            af[i] = load_frag_nk(Ab, rA + 16 * i, lda, kb, lane);
#pragma unroll
        for (int i = 0; i < 4; ++i)
#pragma unroll
            for (int j = 0; j < NT; ++j)
                acc[i][j] = __builtin_amdgcn_wmma_f32_16x16x32_bf16(
                    false, af[i], false, bf[j], (short)0, acc[i][j], false, false);
    }

    const int mo = (lane < 16) ? 0 : 8;   // C/D layout: lanes 16-31 hold M=8..15
#pragma unroll
    for (int j = 0; j < NT; ++j) {
        const int n = cn + 16 * j;
        const float bv = bias ? bias[n] : 0.0f;
#pragma unroll
        for (int i = 0; i < 4; ++i)
#pragma unroll
            for (int r = 0; r < 8; ++r)
                Cb[(size_t)(m0 + 16 * i + mo + r) * ldc + n] = acc[i][j][r] + bv;
    }
}

// ---------------- setup kernels ----------------
__global__ void cvt_f32_bf16(const float* __restrict__ src, uint16_t* __restrict__ dst, long long n) {
    long long i = (long long)blockIdx.x * blockDim.x + threadIdx.x;
    if (i < n) dst[i] = f32_to_bf16(src[i]);
}

__global__ void pack_wcomb(const float* __restrict__ W_ih, const float* __restrict__ W_hh,
                           uint16_t* __restrict__ wc) {
    int i = blockIdx.x * blockDim.x + threadIdx.x;   // 2048*1024
    int n = i >> 10, j = i & 1023;
    float v = (j < 512) ? W_ih[n * 512 + j] : W_hh[n * 512 + (j - 512)];
    wc[i] = f32_to_bf16(v);
}

__global__ void bias_comb(const float* __restrict__ b_ih, const float* __restrict__ b_hh,
                          float* __restrict__ bc) {
    int i = blockIdx.x * blockDim.x + threadIdx.x;   // 2048
    bc[i] = b_ih[i] + b_hh[i];
}

__global__ void init_state(const float* __restrict__ hs0, const float* __restrict__ cs0,
                           float* __restrict__ hs, float* __restrict__ cs) {
    int i = blockIdx.x * blockDim.x + threadIdx.x;   // 64*512
    hs[i] = hs0[i];
    cs[i] = cs0[i];
}

// ---------------- per-step kernels ----------------
__global__ void step_pack(const float* __restrict__ hs, const float* __restrict__ cs,
                          const float* __restrict__ target, int t,
                          uint16_t* __restrict__ xh_df, uint16_t* __restrict__ xh_lstm) {
    int i = blockIdx.x * blockDim.x + threadIdx.x;   // 64*1024
    int b = i >> 10, j = i & 1023;
    float vdf = (j < 512) ? hs[b * 512 + j] : cs[b * 512 + (j - 512)];
    float vl  = (j < 512) ? target[((size_t)b * T_ + t) * E_ + j] : hs[b * 512 + (j - 512)];
    xh_df[i]  = f32_to_bf16(vdf);
    xh_lstm[i] = f32_to_bf16(vl);
}

// one block per batch row; 512 threads = (4 channel-chunks) x (128 positions)
// e[b,s] = sum_c we_w[c]*tanh(conv[b,c*128+s] + df[b,(c&7)*128+s]); softmax;
// h_star = a @ encoder_output[b]
__global__ void attention_kernel(const float* __restrict__ conv, const float* __restrict__ df,
                                 const float* __restrict__ we_w, const float* __restrict__ enc,
                                 uint16_t* __restrict__ feat) {
    const int b = blockIdx.x;
    const int tid = threadIdx.x;       // 0..511
    const int s = tid & 127;
    const int cw = tid >> 7;           // 0..3
    const float* convb = conv + (size_t)b * (A_ * S_);
    const float* dfb = df + (size_t)b * A_;
    float df8[8];
#pragma unroll
    for (int j = 0; j < 8; ++j) df8[j] = dfb[j * 128 + s];

    __shared__ float pe[4][S_];
    __shared__ float sh[S_];

    float part = 0.0f;
    const int c0 = cw * 256;
    for (int c = c0; c < c0 + 256; ++c) {
        float v = convb[c * 128 + s] + df8[c & 7];
        part += we_w[c] * tanhf(v);
    }
    pe[cw][s] = part;
    __syncthreads();
    if (cw == 0) sh[s] = pe[0][s] + pe[1][s] + pe[2][s] + pe[3][s];
    __syncthreads();
    float mx = sh[0];
    for (int i = 1; i < S_; ++i) mx = fmaxf(mx, sh[i]);
    __syncthreads();
    if (cw == 0) sh[s] = __expf(sh[s] - mx);
    __syncthreads();
    float sum = 0.0f;
    for (int i = 0; i < S_; ++i) sum += sh[i];
    const float inv = 1.0f / sum;

    const float* encb = enc + (size_t)b * (S_ * A_);
#pragma unroll
    for (int j = 0; j < 2; ++j) {
        int a = tid + 512 * j;
        float acc = 0.0f;
        for (int si = 0; si < S_; ++si) acc += sh[si] * encb[(size_t)si * A_ + a];
        feat[(size_t)b * 1536 + a] = f32_to_bf16(acc * inv);
    }
}

__global__ void lstm_update(const float* __restrict__ gates, float* __restrict__ hs,
                            float* __restrict__ cs, uint16_t* __restrict__ feat) {
    int i = blockIdx.x * blockDim.x + threadIdx.x;   // 64*512
    int b = i >> 9, j = i & 511;
    const float* g = gates + (size_t)b * 2048;
    float ig = sigm(g[j]);
    float fg = sigm(g[512 + j]);
    float gg = tanhf(g[1024 + j]);
    float og = sigm(g[1536 + j]);
    float cn = fg * cs[i] + ig * gg;
    float hn = og * tanhf(cn);
    cs[i] = cn;
    hs[i] = hn;
    feat[(size_t)b * 1536 + 1024 + j] = f32_to_bf16(hn);
}

__global__ void log_softmax_rows(const float* __restrict__ logits, float* __restrict__ out, int t) {
    const int b = blockIdx.x;
    const int tid = threadIdx.x;  // 256
    const float* row = logits + (size_t)b * VOC_;
    __shared__ float red[256];
    float mx = -INFINITY;
    for (int v = tid; v < VOC_; v += 256) mx = fmaxf(mx, row[v]);
    red[tid] = mx; __syncthreads();
    for (int s = 128; s > 0; s >>= 1) { if (tid < s) red[tid] = fmaxf(red[tid], red[tid + s]); __syncthreads(); }
    mx = red[0]; __syncthreads();
    float sum = 0.0f;
    for (int v = tid; v < VOC_; v += 256) sum += __expf(row[v] - mx);
    red[tid] = sum; __syncthreads();
    for (int s = 128; s > 0; s >>= 1) { if (tid < s) red[tid] += red[tid + s]; __syncthreads(); }
    const float lse = mx + __logf(red[0]);
    float* orow = out + ((size_t)t * B_ + b) * VOC_;
    for (int v = tid; v < VOC_; v += 256) orow[v] = row[v] - lse;
}

// ---------------- workspace layout (bytes, all 256-aligned) ----------------
constexpr size_t OFF_VP   = 0;                          // Vp_w bf16   [32000,1536]
constexpr size_t OFF_WC   = OFF_VP   + 98304000;        // Wcomb bf16  [2048,1024]
constexpr size_t OFF_WSW  = OFF_WC   + 4194304;         // ws_w bf16   [1024,1024]
constexpr size_t OFF_WHW  = OFF_WSW  + 2097152;         // wh_w bf16   [1024,1024]
constexpr size_t OFF_EO   = OFF_WHW  + 2097152;         // enc bf16    [64*128*1024]
constexpr size_t OFF_CONV = OFF_EO   + 16777216;        // conv fp32   [64,1024,128]
constexpr size_t OFF_HS   = OFF_CONV + 33554432;        // hs fp32     [64,512]
constexpr size_t OFF_CS   = OFF_HS   + 131072;          // cs fp32
constexpr size_t OFF_XDF  = OFF_CS   + 131072;          // [hs|cs] bf16 [64,1024]
constexpr size_t OFF_XL   = OFF_XDF  + 131072;          // [x|hs] bf16  [64,1024]
constexpr size_t OFF_DF   = OFF_XL   + 131072;          // df fp32      [64,1024]
constexpr size_t OFF_GT   = OFF_DF   + 262144;          // gates fp32   [64,2048]
constexpr size_t OFF_FEAT = OFF_GT   + 524288;          // feat bf16    [64,1536]
constexpr size_t OFF_LG   = OFF_FEAT + 196608;          // logits fp32  [64,32000]
constexpr size_t OFF_BC   = OFF_LG   + 8192000;         // b_ih+b_hh fp32 [2048]

extern "C" void kernel_launch(void* const* d_in, const int* in_sizes, int n_in,
                              void* d_out, int out_size, void* d_ws, size_t ws_size,
                              hipStream_t stream) {
    const float* enc    = (const float*)d_in[0];
    const float* hs0    = (const float*)d_in[1];
    const float* cs0    = (const float*)d_in[2];
    const float* target = (const float*)d_in[3];
    const float* wh_w   = (const float*)d_in[4];
    const float* ws_w   = (const float*)d_in[5];
    const float* ws_b   = (const float*)d_in[6];
    const float* we_w   = (const float*)d_in[7];
    const float* W_ih   = (const float*)d_in[8];
    const float* W_hh   = (const float*)d_in[9];
    const float* b_ih   = (const float*)d_in[10];
    const float* b_hh   = (const float*)d_in[11];
    const float* Vp_w   = (const float*)d_in[12];
    const float* Vp_b   = (const float*)d_in[13];
    float* out = (float*)d_out;
    char* w = (char*)d_ws;

    uint16_t* vp_bf   = (uint16_t*)(w + OFF_VP);
    uint16_t* wc_bf   = (uint16_t*)(w + OFF_WC);
    uint16_t* wsw_bf  = (uint16_t*)(w + OFF_WSW);
    uint16_t* whw_bf  = (uint16_t*)(w + OFF_WHW);
    uint16_t* eo_bf   = (uint16_t*)(w + OFF_EO);
    float*    conv    = (float*)   (w + OFF_CONV);
    float*    hs      = (float*)   (w + OFF_HS);
    float*    cs      = (float*)   (w + OFF_CS);
    uint16_t* xh_df   = (uint16_t*)(w + OFF_XDF);
    uint16_t* xh_l    = (uint16_t*)(w + OFF_XL);
    float*    df      = (float*)   (w + OFF_DF);
    float*    gates   = (float*)   (w + OFF_GT);
    uint16_t* feat    = (uint16_t*)(w + OFF_FEAT);
    float*    logits  = (float*)   (w + OFF_LG);
    float*    bc      = (float*)   (w + OFF_BC);

    // ---- one-time conversions / packs (re-run each call; deterministic) ----
    {
        long long n = (long long)VOC_ * 1536;
        cvt_f32_bf16<<<dim3((unsigned)((n + 255) / 256)), 256, 0, stream>>>(Vp_w, vp_bf, n);
    }
    pack_wcomb<<<dim3(2048 * 1024 / 256), 256, 0, stream>>>(W_ih, W_hh, wc_bf);
    cvt_f32_bf16<<<dim3(1024 * 1024 / 256), 256, 0, stream>>>(ws_w, wsw_bf, 1024LL * 1024);
    cvt_f32_bf16<<<dim3(1024 * 1024 / 256), 256, 0, stream>>>(wh_w, whw_bf, 1024LL * 1024);
    cvt_f32_bf16<<<dim3(B_ * S_ * A_ / 256), 256, 0, stream>>>(enc, eo_bf, (long long)B_ * S_ * A_);
    bias_comb<<<dim3(2048 / 256), 256, 0, stream>>>(b_ih, b_hh, bc);
    init_state<<<dim3(B_ * H_ / 256), 256, 0, stream>>>(hs0, cs0, hs, cs);

    // conv[b] = wh_w (MxK=1024x1024) x eo[b] (KxN=1024x128), batched over B.
    // NT=1: each wave one 16-col tile (N=128 -> 2 blocks of 4 waves).
    gemm_wmma_bf16<true, 1><<<dim3(S_ / 64, A_ / 64, B_), 128, 0, stream>>>(
        whw_bf, 0LL, A_, eo_bf, (long long)A_ * S_, S_, nullptr,
        conv, (long long)A_ * S_, S_, S_, A_);

    // ---- timestep loop ----
    for (int t = 0; t < T_; ++t) {
        step_pack<<<dim3(B_ * A_ / 256), 256, 0, stream>>>(hs, cs, target, t, xh_df, xh_l);

        // df = [hs|cs] @ ws_w^T + ws_b   (M=64, N=1024, K=1024); 256 cols/block
        gemm_wmma_bf16<false, 4><<<dim3(A_ / 256, 1, 1), 128, 0, stream>>>(
            xh_df, 0LL, A_, wsw_bf, 0LL, A_, ws_b, df, 0LL, A_, A_, A_);

        attention_kernel<<<dim3(B_), 512, 0, stream>>>(conv, df, we_w, enc, feat);

        // gates = [x_t|hs] @ [W_ih|W_hh]^T + (b_ih+b_hh)  (M=64, N=2048, K=1024)
        gemm_wmma_bf16<false, 4><<<dim3(2048 / 256, 1, 1), 128, 0, stream>>>(
            xh_l, 0LL, A_, wc_bf, 0LL, A_, bc, gates, 0LL, 2048, 2048, A_);

        lstm_update<<<dim3(B_ * H_ / 256), 256, 0, stream>>>(gates, hs, cs, feat);

        // logits = feat @ Vp_w^T + Vp_b  (M=64, N=32000, K=1536); 125 blocks
        gemm_wmma_bf16<false, 4><<<dim3(VOC_ / 256, 1, 1), 128, 0, stream>>>(
            feat, 0LL, 1536, vp_bf, 0LL, 1536, Vp_b, logits, 0LL, VOC_, VOC_, 1536);

        log_softmax_rows<<<dim3(B_), 256, 0, stream>>>(logits, out, t);
    }
}
